// SocialForceGNN_24567212934012
// MI455X (gfx1250) — compile-verified, matrix operand order checked by reading, not run on hardware
//
#include <hip/hip_runtime.h>
#include <hip/hip_bf16.h>

#define N_NODES 50000
#define N_EDGES 800000
#define N_GRAPHS 50
#define HDIM 128

typedef __bf16 bf16_t;
typedef __bf16 v16bf __attribute__((ext_vector_type(16)));
typedef float  v8f   __attribute__((ext_vector_type(8)));
typedef unsigned int u32x4 __attribute__((ext_vector_type(4)));
typedef unsigned int u32x2 __attribute__((ext_vector_type(2)));

// ---------------------------------------------------------------------------
// CDNA5 async global->LDS copy (tracked on ASYNCcnt), per-lane 8 bytes.
// LDS byte offset = low 32 bits of the generic (flat) address of a __shared__
// object (LDS aperture keeps the wave-relative LDS address in addr[31:0]).
// ---------------------------------------------------------------------------
__device__ __forceinline__ void async_copy_b64(unsigned lds_off, const void* gptr) {
    unsigned long long ga = (unsigned long long)(uintptr_t)gptr;
    asm volatile("global_load_async_to_lds_b64 %0, %1, off"
                 :
                 : "v"(lds_off), "v"(ga)
                 : "memory");
}
__device__ __forceinline__ void wait_async_zero() {
    asm volatile("s_wait_asynccnt 0" ::: "memory");
}

// ---------------------------------------------------------------------------
// WMMA fragment helpers (wave32, 16x16x32 bf16, D/C = f32)
//
// A (16x32 bf16, row-major source with leading dim ldk):
//   lane L: m = L%16, k0 = (L<16 ? 0 : 8) + 32*kc
//   elems e[0..7]  = A[m][k0+0 .. k0+7]
//   elems e[8..15] = A[m][k0+16 .. k0+23]
// B is pre-swizzled on device into per-lane 32-byte fragments so the load is
// two contiguous b128s: Wswz[((kc*8 + t)*32 + lane)*16 + j].
// D/C (16x16 f32): lane L: n = L%16, vgpr r -> m = r + (L<16 ? 0 : 8).
// ---------------------------------------------------------------------------

__device__ __forceinline__ v16bf load_afrag(const bf16_t* A, int ldk, int kc, int lane) {
    int m  = lane & 15;
    int k0 = kc * 32 + ((lane & 16) ? 8 : 0);
    union { v16bf v; u32x4 q[2]; } u;
    u.q[0] = *(const u32x4*)(A + m * ldk + k0);
    u.q[1] = *(const u32x4*)(A + m * ldk + k0 + 16);
    return u.v;
}

__device__ __forceinline__ v16bf load_bfrag(const bf16_t* Wswz, int kc, int t, int lane) {
    union { v16bf v; u32x4 q[2]; } u;
    const u32x4* p = (const u32x4*)(Wswz + (size_t)(((kc * 8 + t) * 32 + lane) * 16));
    u.q[0] = p[0];
    u.q[1] = p[1];
    return u.v;
}

// C[16x128] += A[16x(NKC*32)] * W[(NKC*32)x128], accumulators in 8 col tiles.
template <int NKC>
__device__ __forceinline__ void gemm_tile(const bf16_t* A, int ldk, const bf16_t* Wswz,
                                          int lane, v8f acc[8]) {
#pragma unroll
    for (int kc = 0; kc < NKC; ++kc) {
        v16bf a = load_afrag(A, ldk, kc, lane);
#pragma unroll
        for (int t = 0; t < 8; ++t) {
            v16bf b = load_bfrag(Wswz, kc, t, lane);
            acc[t] = __builtin_amdgcn_wmma_f32_16x16x32_bf16(
                false, a, false, b, (short)0, acc[t], false, false);
        }
    }
}

// relu(acc + b1) -> LDS hidden tile (16x128 bf16, row-major)
__device__ __forceinline__ void store_hidden_relu(bf16_t* Hl, const v8f acc[8],
                                                  const float* b1, int lane) {
    int n0    = lane & 15;
    int mbase = (lane & 16) ? 8 : 0;
#pragma unroll
    for (int t = 0; t < 8; ++t) {
        int   col  = t * 16 + n0;
        float bias = b1[col];
#pragma unroll
        for (int r = 0; r < 8; ++r) {
            float v = acc[t][r] + bias;
            v       = v > 0.f ? v : 0.f;
            Hl[(mbase + r) * HDIM + col] = (bf16_t)v;
        }
    }
}

// ---------------------------------------------------------------------------
// Weight swizzle: fp32 row-major W[K][128] -> bf16 per-lane WMMA B fragments
// ---------------------------------------------------------------------------
__global__ void swz_weight_kernel(const float* __restrict__ W, bf16_t* __restrict__ out, int K) {
    int idx   = blockIdx.x * blockDim.x + threadIdx.x;  // one thread per (kc,t,lane)
    int total = (K / 32) * 8 * 32;
    if (idx >= total) return;
    int lane = idx & 31;
    int t    = (idx >> 5) & 7;
    int kc   = idx >> 8;
    int n    = (lane & 15) + t * 16;
    int k0   = kc * 32 + ((lane & 16) ? 8 : 0);
    bf16_t* o = out + (size_t)idx * 16;
#pragma unroll
    for (int j = 0; j < 8; ++j) {
        o[j]     = (bf16_t)W[(k0 + j) * HDIM + n];
        o[8 + j] = (bf16_t)W[(k0 + 16 + j) * HDIM + n];
    }
}

// ---------------------------------------------------------------------------
// Utility kernels
// ---------------------------------------------------------------------------
__global__ void zero_f32_kernel(float* p, int n) {
    for (int i = blockIdx.x * blockDim.x + threadIdx.x; i < n; i += gridDim.x * blockDim.x)
        p[i] = 0.f;
}

__global__ void deg_kernel(const int* __restrict__ dst, float* __restrict__ deg) {
    int i = blockIdx.x * blockDim.x + threadIdx.x;
    if (i < N_EDGES) atomicAdd(&deg[dst[i]], 1.f);
}

__global__ void invdeg_kernel(float* deg) {
    int i = blockIdx.x * blockDim.x + threadIdx.x;
    if (i < N_NODES) {
        float d = deg[i];
        d       = d > 1.f ? d : 1.f;
        deg[i]  = 1.f / d;  // in-place: buffer becomes invdeg
    }
}

// ---------------------------------------------------------------------------
// Encoder: out = (relu(X @ W1 + b1)) @ W2 + b2 ; X is [nrows x Kin], Kin tiny.
// One wave per 16-row tile; layer1 in VALU, layer2 in WMMA.
// ---------------------------------------------------------------------------
__global__ __launch_bounds__(32) void encode_kernel(
    const float* __restrict__ X, int Kin, int nrows,
    const float* __restrict__ W1, const float* __restrict__ b1,
    const bf16_t* __restrict__ W2s, const float* __restrict__ b2,
    float* __restrict__ out_f32, bf16_t* __restrict__ out_bf16) {
    __shared__ __align__(16) bf16_t Hl[16 * HDIM];
    int lane = threadIdx.x;
    int row0 = blockIdx.x * 16;
    __builtin_prefetch(W2s, 0, 0);

    for (int i = lane; i < 16 * HDIM; i += 32) {
        int m = i >> 7, j = i & 127;
        int r = row0 + m;
        r     = r < nrows ? r : nrows - 1;
        float s = b1[j];
        for (int k = 0; k < Kin; ++k) s += X[(size_t)r * Kin + k] * W1[k * HDIM + j];
        s = s > 0.f ? s : 0.f;
        Hl[m * HDIM + j] = (bf16_t)s;
    }
    __syncthreads();

    v8f acc[8] = {};
    gemm_tile<4>(Hl, HDIM, W2s, lane, acc);

    int n0    = lane & 15;
    int mbase = (lane & 16) ? 8 : 0;
#pragma unroll
    for (int t = 0; t < 8; ++t) {
        int   col  = t * 16 + n0;
        float bias = b2[col];
#pragma unroll
        for (int r = 0; r < 8; ++r) {
            int row = row0 + mbase + r;
            if (row < nrows) {
                float v = acc[t][r] + bias;
                if (out_f32) out_f32[(size_t)row * HDIM + col] = v;
                out_bf16[(size_t)row * HDIM + col] = (bf16_t)v;
            }
        }
    }
}

// ---------------------------------------------------------------------------
// Message MLP + scaled atomic aggregation. One wave per 16-edge tile.
//   m = mlp2([h[src] | h[dst] | e]) ; aggr[dst] += m * invdeg[dst]
// Staging of the 16x384 bf16 input tile uses CDNA5 async global->LDS copies.
// ---------------------------------------------------------------------------
__global__ __launch_bounds__(32) void msg_kernel(
    const bf16_t* __restrict__ hbf, const bf16_t* __restrict__ ebf,
    const int* __restrict__ src, const int* __restrict__ dst,
    const bf16_t* __restrict__ W1s, const float* __restrict__ b1,
    const bf16_t* __restrict__ W2s, const float* __restrict__ b2,
    const float* __restrict__ invdeg, float* __restrict__ aggr) {
    __shared__ __align__(16) bf16_t Ain[16 * 384];
    __shared__ __align__(16) bf16_t Hl[16 * HDIM];
    __shared__ int   dsts[16];
    __shared__ float invw[16];

    int lane = threadIdx.x;
    int e0   = blockIdx.x * 16;
    __builtin_prefetch(W1s, 0, 0);
    __builtin_prefetch(W2s, 0, 0);

    if (lane < 16) {
        int d      = dst[e0 + lane];
        dsts[lane] = d;
        invw[lane] = invdeg[d];
    }
    __syncthreads();

    // Stage [h_src | h_dst | e] as 16 rows x 384 bf16 via async-to-LDS:
    // each lane moves 8B per row per segment, no VGPR round-trip.
    unsigned abase = (unsigned)(uintptr_t)(&Ain[0]);
    for (int r = 0; r < 16; ++r) {
        int e = e0 + r;
        int s = src[e];
        int d = dsts[r];
        async_copy_b64(abase + (unsigned)((r * 384 + lane * 4) * 2),
                       &hbf[(size_t)s * HDIM + lane * 4]);
        async_copy_b64(abase + (unsigned)((r * 384 + 128 + lane * 4) * 2),
                       &hbf[(size_t)d * HDIM + lane * 4]);
        async_copy_b64(abase + (unsigned)((r * 384 + 256 + lane * 4) * 2),
                       &ebf[(size_t)e * HDIM + lane * 4]);
    }
    wait_async_zero();
    __syncthreads();

    v8f acc[8] = {};
    gemm_tile<12>(Ain, 384, W1s, lane, acc);
    store_hidden_relu(Hl, acc, b1, lane);
    __syncthreads();

    v8f acc2[8] = {};
    gemm_tile<4>(Hl, HDIM, W2s, lane, acc2);

    int n0    = lane & 15;
    int mbase = (lane & 16) ? 8 : 0;
#pragma unroll
    for (int t = 0; t < 8; ++t) {
        int   col  = t * 16 + n0;
        float bias = b2[col];
#pragma unroll
        for (int r = 0; r < 8; ++r) {
            int   m = mbase + r;
            float v = (acc2[t][r] + bias) * invw[m];
            atomicAdd(&aggr[(size_t)dsts[m] * HDIM + col], v);
        }
    }
}

// ---------------------------------------------------------------------------
// Update MLP + residual: h += mlp2([h | aggr]); also refresh bf16 h, zero aggr.
// ---------------------------------------------------------------------------
__global__ __launch_bounds__(32) void upd_kernel(
    float* __restrict__ h, bf16_t* __restrict__ hbf, float* __restrict__ aggr,
    const bf16_t* __restrict__ W1s, const float* __restrict__ b1,
    const bf16_t* __restrict__ W2s, const float* __restrict__ b2) {
    __shared__ __align__(16) bf16_t Ain[16 * 256];
    __shared__ __align__(16) bf16_t Hl[16 * HDIM];

    int lane = threadIdx.x;
    int row0 = blockIdx.x * 16;
    __builtin_prefetch(W1s, 0, 0);

    for (int r = 0; r < 16; ++r) {
        int    row = row0 + r;
        float4 hv  = *(const float4*)&h[(size_t)row * HDIM + lane * 4];
        float4 av  = *(const float4*)&aggr[(size_t)row * HDIM + lane * 4];
        bf16_t* ph = &Ain[r * 256 + lane * 4];
        ph[0] = (bf16_t)hv.x; ph[1] = (bf16_t)hv.y;
        ph[2] = (bf16_t)hv.z; ph[3] = (bf16_t)hv.w;
        bf16_t* pa = &Ain[r * 256 + 128 + lane * 4];
        pa[0] = (bf16_t)av.x; pa[1] = (bf16_t)av.y;
        pa[2] = (bf16_t)av.z; pa[3] = (bf16_t)av.w;
        // Re-zero aggr for the next layer / next graph replay.
        *(float4*)&aggr[(size_t)row * HDIM + lane * 4] = make_float4(0.f, 0.f, 0.f, 0.f);
    }
    __syncthreads();

    v8f acc[8] = {};
    gemm_tile<8>(Ain, 256, W1s, lane, acc);
    store_hidden_relu(Hl, acc, b1, lane);
    __syncthreads();

    v8f acc2[8] = {};
    gemm_tile<4>(Hl, HDIM, W2s, lane, acc2);

    int n0    = lane & 15;
    int mbase = (lane & 16) ? 8 : 0;
#pragma unroll
    for (int t = 0; t < 8; ++t) {
        int   col  = t * 16 + n0;
        float bias = b2[col];
#pragma unroll
        for (int r = 0; r < 8; ++r) {
            int    row = row0 + mbase + r;
            size_t ix  = (size_t)row * HDIM + col;
            float  nh  = h[ix] + acc2[t][r] + bias;
            h[ix]      = nh;
            hbf[ix]    = (bf16_t)nh;
        }
    }
}

// ---------------------------------------------------------------------------
// Head: out[g] = mlp2(h[ptr[g]]) -> 2 floats per graph (tiny, scalar VALU)
// ---------------------------------------------------------------------------
__global__ __launch_bounds__(128) void head_kernel(
    const float* __restrict__ h, const int* __restrict__ ptr,
    const float* __restrict__ W1, const float* __restrict__ b1,
    const float* __restrict__ W2, const float* __restrict__ b2,
    float* __restrict__ out) {
    __shared__ float hr[HDIM];
    __shared__ float hid[HDIM];
    int g = blockIdx.x, t = threadIdx.x;
    int node = ptr[g];
    hr[t] = h[(size_t)node * HDIM + t];
    __syncthreads();
    float s = b1[t];
    for (int k = 0; k < HDIM; ++k) s += hr[k] * W1[k * HDIM + t];
    hid[t] = s > 0.f ? s : 0.f;
    __syncthreads();
    if (t < 2) {
        float o = b2[t];
        for (int k = 0; k < HDIM; ++k) o += hid[k] * W2[k * 2 + t];
        out[g * 2 + t] = o;
    }
}

// ---------------------------------------------------------------------------
// Host launch
// ---------------------------------------------------------------------------
extern "C" void kernel_launch(void* const* d_in, const int* in_sizes, int n_in,
                              void* d_out, int out_size, void* d_ws, size_t ws_size,
                              hipStream_t stream) {
    (void)in_sizes; (void)n_in; (void)out_size; (void)ws_size;

    const float* x         = (const float*)d_in[0];
    const float* edge_attr = (const float*)d_in[1];
    const int*   eidx      = (const int*)d_in[2];   // harness passes integers as int32
    const int*   srcp      = eidx;
    const int*   dstp      = eidx + N_EDGES;
    const int*   ptrp      = (const int*)d_in[3];

    // params flattened in dict insertion order starting at d_in[4]
    auto P = [&](int i) { return (const float*)d_in[4 + i]; };
    const float* node_W1 = P(0); const float* node_b1 = P(1);
    const float* node_W2 = P(2); const float* node_b2 = P(3);
    const float* edge_W1 = P(4); const float* edge_b1 = P(5);
    const float* edge_W2 = P(6); const float* edge_b2 = P(7);
    const float* msg_W1[3]; const float* msg_b1[3]; const float* msg_W2[3]; const float* msg_b2[3];
    const float* upd_W1[3]; const float* upd_b1[3]; const float* upd_W2[3]; const float* upd_b2[3];
    for (int l = 0; l < 3; ++l) {
        msg_W1[l] = P(8 + 8 * l);  msg_b1[l] = P(9 + 8 * l);
        msg_W2[l] = P(10 + 8 * l); msg_b2[l] = P(11 + 8 * l);
        upd_W1[l] = P(12 + 8 * l); upd_b1[l] = P(13 + 8 * l);
        upd_W2[l] = P(14 + 8 * l); upd_b2[l] = P(15 + 8 * l);
    }
    const float* head_W1 = P(32); const float* head_b1 = P(33);
    const float* head_W2 = P(34); const float* head_b2 = P(35);

    // Workspace carve-out
    size_t off = 0;
    auto carve = [&](size_t bytes) {
        void* p = (char*)d_ws + off;
        off += (bytes + 255) & ~(size_t)255;
        return p;
    };
    float*  h    = (float*)carve((size_t)N_NODES * HDIM * 4);
    bf16_t* hbf  = (bf16_t*)carve((size_t)N_NODES * HDIM * 2);
    bf16_t* ebf  = (bf16_t*)carve((size_t)N_EDGES * HDIM * 2);
    float*  aggr = (float*)carve((size_t)N_NODES * HDIM * 4);
    float*  deg  = (float*)carve((size_t)N_NODES * 4);  // becomes invdeg in-place
    bf16_t* nodeW2s = (bf16_t*)carve(128 * 128 * 2);
    bf16_t* edgeW2s = (bf16_t*)carve(128 * 128 * 2);
    bf16_t* msgW1s[3]; bf16_t* msgW2s[3]; bf16_t* updW1s[3]; bf16_t* updW2s[3];
    for (int l = 0; l < 3; ++l) {
        msgW1s[l] = (bf16_t*)carve(384 * 128 * 2);
        msgW2s[l] = (bf16_t*)carve(128 * 128 * 2);
        updW1s[l] = (bf16_t*)carve(256 * 128 * 2);
        updW2s[l] = (bf16_t*)carve(128 * 128 * 2);
    }

    // 1) zero accumulation buffers (every call -> deterministic replay)
    zero_f32_kernel<<<512, 256, 0, stream>>>(aggr, N_NODES * HDIM);
    zero_f32_kernel<<<64, 256, 0, stream>>>(deg, N_NODES);

    // 2) weight swizzle to bf16 WMMA fragments
    struct Job { const float* W; bf16_t* out; int K; };
    Job jobs[14] = {
        {node_W2, nodeW2s, 128}, {edge_W2, edgeW2s, 128},
        {msg_W1[0], msgW1s[0], 384}, {msg_W2[0], msgW2s[0], 128},
        {upd_W1[0], updW1s[0], 256}, {upd_W2[0], updW2s[0], 128},
        {msg_W1[1], msgW1s[1], 384}, {msg_W2[1], msgW2s[1], 128},
        {upd_W1[1], updW1s[1], 256}, {upd_W2[1], updW2s[1], 128},
        {msg_W1[2], msgW1s[2], 384}, {msg_W2[2], msgW2s[2], 128},
        {upd_W1[2], updW1s[2], 256}, {upd_W2[2], updW2s[2], 128},
    };
    for (int j = 0; j < 14; ++j) {
        int total = (jobs[j].K / 32) * 8 * 32;
        swz_weight_kernel<<<(total + 255) / 256, 256, 0, stream>>>(jobs[j].W, jobs[j].out, jobs[j].K);
    }

    // 3) degree -> invdeg
    deg_kernel<<<(N_EDGES + 255) / 256, 256, 0, stream>>>(dstp, deg);
    invdeg_kernel<<<(N_NODES + 255) / 256, 256, 0, stream>>>(deg);

    // 4) encoders
    encode_kernel<<<N_NODES / 16, 32, 0, stream>>>(x, 5, N_NODES, node_W1, node_b1,
                                                   nodeW2s, node_b2, h, hbf);
    encode_kernel<<<N_EDGES / 16, 32, 0, stream>>>(edge_attr, 7, N_EDGES, edge_W1, edge_b1,
                                                   edgeW2s, edge_b2, nullptr, ebf);

    // 5) message-passing layers
    for (int l = 0; l < 3; ++l) {
        msg_kernel<<<N_EDGES / 16, 32, 0, stream>>>(hbf, ebf, srcp, dstp,
                                                    msgW1s[l], msg_b1[l],
                                                    msgW2s[l], msg_b2[l], deg, aggr);
        upd_kernel<<<N_NODES / 16, 32, 0, stream>>>(h, hbf, aggr,
                                                    updW1s[l], upd_b1[l],
                                                    updW2s[l], upd_b2[l]);
    }

    // 6) head
    head_kernel<<<N_GRAPHS, 128, 0, stream>>>(h, ptrp, head_W1, head_b1, head_W2, head_b2,
                                              (float*)d_out);
}
